// MPS_BP_40638980555109
// MI455X (gfx1250) — matrix operand order repaired and checked by reading, not compiled, not run on hardware
//
#include <hip/hip_runtime.h>

#define BATCH   1024
#define CHANNEL 10
#define LENGTH  784
#define CHI     64
#define DD      2
#define MTILE   64
#define AST     72   // padded LDS row stride (u16 elems); multiple of 8 keeps 16B frag alignment

typedef __attribute__((ext_vector_type(16))) __bf16 v16bf;
typedef __attribute__((ext_vector_type(8)))  float  v8f;

union Frag { v16bf v; uint4 q[2]; };

template<bool B> struct BoolC { static constexpr bool value = B; };

__device__ __forceinline__ unsigned short f2bf(float f) {
  unsigned int u = __float_as_uint(f);
  u += 0x7FFFu + ((u >> 16) & 1u);     // round-to-nearest-even
  return (unsigned short)(u >> 16);
}

// pack two f32 -> two bf16 in one dword (one v_cvt_pk_bf16_f32 when available)
__device__ __forceinline__ unsigned pk_bf16(float lo, float hi) {
#if __has_builtin(__builtin_amdgcn_cvt_pk_bf16_f32)
  typedef __attribute__((ext_vector_type(2))) __bf16 v2bf;
  v2bf p = __builtin_amdgcn_cvt_pk_bf16_f32(lo, hi);
  unsigned u;
  __builtin_memcpy(&u, &p, sizeof(u));
  return u;
#else
  return (unsigned)f2bf(lo) | ((unsigned)f2bf(hi) << 16);
#endif
}

// (tanh(u)+1)/2 == 1/(1+exp(-2u)) : branchless, saturates correctly at both ends
__device__ __forceinline__ float act_sigmoid2(float u) {
  float e = __builtin_amdgcn_exp2f(u * -2.8853900817779268f); // -2*log2(e)
  return __builtin_amdgcn_rcpf(1.0f + e);
}

// dims[k] = min(2^k, 64, 2^(784-k)); dims[0] = 1
__device__ __forceinline__ int bond_dim(int k) {
  if (k <= 0) return 1;
  int a = (k >= 6) ? 64 : (1 << k);
  int r = LENGTH - k;
  int b = (r >= 6) ? 64 : (1 << r);
  return a < b ? a : b;
}

__global__ __launch_bounds__(256)
void mps_fwd(const float* __restrict__ x, const float* __restrict__ tensors,
             float* __restrict__ out) {
  // B-operand staging: layout [i*64 + b][a], a contiguous -> contiguous-K lane reads
  __shared__ __align__(16) unsigned short ldsA[128 * AST];   // 18 KB
  __shared__ __align__(16) unsigned short ldsY[MTILE * AST]; //  9 KB, state y[m][k]
  __shared__ float ldsX[MTILE * 2];

  const int tid   = threadIdx.x;
  const int lane  = tid & 31;
  const int wave  = tid >> 5;      // 8 wave32s
  const int laneN = lane & 15;
  const int hi    = lane >> 4;     // half-wave select
  const int mtile = wave >> 1;     // 0..3 : 16-row batch tile
  const int bh    = wave & 1;      // 0..1 : which pair of b-blocks
  const int c     = blockIdx.y;
  const int m0    = blockIdx.x * MTILE;

  // init state: y[m][0] = 1.0 (bf16 0x3F80), else 0
  for (int e = tid; e < MTILE * AST; e += 256)
    ldsY[e] = ((e % AST) == 0) ? (unsigned short)0x3F80 : (unsigned short)0;

  const float* gA = tensors + (size_t)c * LENGTH * (CHI * DD * CHI);

  auto step = [&](int site, auto lastc) {
    constexpr bool LAST = decltype(lastc)::value;
    const float4* gAs4 = (const float4*)(gA + (size_t)site * (CHI * DD * CHI));
    const int dN  = bond_dim(site);
    const int dN1 = bond_dim(site + 1);

    // ---- stage site tensor: pairs (a, a+1) -> packed b32 LDS stores ----
    // fast compare-free path for the 772 full-bond sites
    if (dN == CHI && dN1 == CHI) {
      #pragma unroll
      for (int rr = 0; rr < 4; ++rr) {
        int p  = rr * 256 + tid;              // pair index
        int q0 = ((p >> 5) << 6) + (p & 31);  // float4 idx, 'a' even
        float4 v0 = gAs4[q0];                 // element a
        float4 v1 = gAs4[q0 + 32];            // element a+1 (same i,b range)
        int e0  = q0 * 4;
        int a   = e0 >> 7;                    // even chi row
        int row = e0 & 127;                   // i*64 + b (multiple of 4)
        *(unsigned*)(ldsA + (row + 0) * AST + a) = pk_bf16(v0.x, v1.x);
        *(unsigned*)(ldsA + (row + 1) * AST + a) = pk_bf16(v0.y, v1.y);
        *(unsigned*)(ldsA + (row + 2) * AST + a) = pk_bf16(v0.z, v1.z);
        *(unsigned*)(ldsA + (row + 3) * AST + a) = pk_bf16(v0.w, v1.w);
      }
    } else {
      #pragma unroll
      for (int rr = 0; rr < 4; ++rr) {
        int p  = rr * 256 + tid;
        int q0 = ((p >> 5) << 6) + (p & 31);
        float4 v0 = gAs4[q0];
        float4 v1 = gAs4[q0 + 32];
        int e0  = q0 * 4;
        int a   = e0 >> 7;
        int row = e0 & 127;
        int b   = row & 63;
        float f0[4] = {v0.x, v0.y, v0.z, v0.w};
        float f1[4] = {v1.x, v1.y, v1.z, v1.w};
        #pragma unroll
        for (int j = 0; j < 4; ++j) {
          float lo = (a     < dN && (b + j) < dN1) ? f0[j] : 0.0f;
          float hv = (a + 1 < dN && (b + j) < dN1) ? f1[j] : 0.0f;
          *(unsigned*)(ldsA + (row + j) * AST + a) = pk_bf16(lo, hv);
        }
      }
    }
    // stage x[n, i, site]
    if (tid < MTILE * 2) {
      int m = tid >> 1, i = tid & 1;
      ldsX[tid] = x[(size_t)(m0 + m) * (DD * LENGTH) + i * LENGTH + site];
    }
    // prefetch next site's 32KB tensor tile while we compute this one
    if constexpr (!LAST)
      __builtin_prefetch((const float*)gAs4 + (CHI * DD * CHI) + tid * 32, 0, 0);

    __syncthreads();

    // ---- A fragments (state y), two K-blocks; ISA 16-bit A layout ----
    Frag afr[2];
    #pragma unroll
    for (int kb = 0; kb < 2; ++kb) {
      int row = mtile * 16 + laneN;              // M = lane&15
      int k0  = kb * 32 + 8 * hi;                // K chunk base per half-wave
      afr[kb].q[0] = *(const uint4*)(ldsY + row * AST + k0);
      afr[kb].q[1] = *(const uint4*)(ldsY + row * AST + k0 + 16);
    }

    // per-lane x scalars in registers
    float xr[2][8];
    #pragma unroll
    for (int r = 0; r < 8; ++r) {
      int m = mtile * 16 + r + 8 * hi;           // C/D layout: M = r + 8*hi
      xr[0][r] = ldsX[m * 2 + 0];
      xr[1][r] = ldsX[m * 2 + 1];
    }

    // ---- z_i = Y · A_i : 4 accumulators x 2 K-step WMMAs ----
    v8f acc[2][2];                               // [i][b-block]
    #pragma unroll
    for (int i = 0; i < 2; ++i) {
      #pragma unroll
      for (int bb = 0; bb < 2; ++bb) {
        v8f cacc = {0.f, 0.f, 0.f, 0.f, 0.f, 0.f, 0.f, 0.f};
        #pragma unroll
        for (int kb = 0; kb < 2; ++kb) {
          Frag bfr;
          int row = i * 64 + (bh * 2 + bb) * 16 + laneN;  // N = lane&15
          int a0  = kb * 32 + 8 * hi;                     // K chunk base per half-wave
          bfr.q[0] = *(const uint4*)(ldsA + row * AST + a0);
          bfr.q[1] = *(const uint4*)(ldsA + row * AST + a0 + 16);
          cacc = __builtin_amdgcn_wmma_f32_16x16x32_bf16(
              false, afr[kb].v, false, bfr.v, (short)0, cacc, false, false);
        }
        acc[i][bb] = cacc;
      }
    }

    if constexpr (!LAST) {
      __syncthreads();   // all ldsY/ldsA/ldsX reads done before overwrite
      // ---- y' = sigmoid2(x0*z0 + x1*z1), write state back ----
      #pragma unroll
      for (int bb = 0; bb < 2; ++bb) {
        #pragma unroll
        for (int r = 0; r < 8; ++r) {
          int m = mtile * 16 + r + 8 * hi;
          float v = act_sigmoid2(xr[0][r] * acc[0][bb][r] + xr[1][r] * acc[1][bb][r]);
          ldsY[m * AST + (bh * 2 + bb) * 16 + laneN] = f2bf(v);
        }
      }
      // next iteration's first __syncthreads orders state-write -> state-read
    } else {
      // final site: no activation; only b==0 survives (dims[L]==1)
      if (bh == 0 && laneN == 0) {
        #pragma unroll
        for (int r = 0; r < 8; ++r) {
          int m = mtile * 16 + r + 8 * hi;
          float v = xr[0][r] * acc[0][0][r] + xr[1][r] * acc[1][0][r];
          out[(size_t)(m0 + m) * CHANNEL + c] = v;   // y[:, :, 0]
        }
      }
    }
  };

  #pragma unroll 1
  for (int site = 0; site < LENGTH - 1; ++site)
    step(site, BoolC<false>{});
  step(LENGTH - 1, BoolC<true>{});
}

extern "C" void kernel_launch(void* const* d_in, const int* in_sizes, int n_in,
                              void* d_out, int out_size, void* d_ws, size_t ws_size,
                              hipStream_t stream) {
  (void)in_sizes; (void)n_in; (void)d_ws; (void)ws_size; (void)out_size;
  const float* x       = (const float*)d_in[0];  // (1024, 2, 784) f32
  const float* tensors = (const float*)d_in[1];  // (10, 784, 64, 2, 64) f32
  float* out = (float*)d_out;                    // (1024, 10, 1) f32
  dim3 grid(BATCH / MTILE, CHANNEL);
  mps_fwd<<<grid, dim3(256), 0, stream>>>(x, tensors, out);
}
